// HashMemory_70781061038578
// MI455X (gfx1250) — compile-verified, matrix operand order checked by reading, not run on hardware
//
#include <hip/hip_runtime.h>
#include <math.h>

typedef __attribute__((ext_vector_type(16))) _Float16 v16h;
typedef __attribute__((ext_vector_type(8)))  _Float16 v8h;
typedef __attribute__((ext_vector_type(4)))  _Float16 v4h;
typedef __attribute__((ext_vector_type(8)))  float    v8f;

// ---------------------------------------------------------------------------
// WMMA 16x16x32 f16 fragment: lane holds row/col = lane%16; with
// klo = (lane>=16)?8:0 the 16 K-elements are two contiguous 8-half runs at
// +0..7 and +16..23 from (k0+klo).  Two global_load_b128.
// ---------------------------------------------------------------------------
static __device__ inline v16h frag16(const _Float16* __restrict__ p) {
  v8h lo = *(const v8h*)(p);
  v8h hi = *(const v8h*)(p + 16);
  return __builtin_shufflevector(lo, hi, 0, 1, 2, 3, 4, 5, 6, 7,
                                 8, 9, 10, 11, 12, 13, 14, 15);
}

// ---------------------------------------------------------------------------
// Pre-pass 1: f32 -> f16 elementwise
// ---------------------------------------------------------------------------
__global__ __launch_bounds__(256) void hm_cvt_f16(
    const float* __restrict__ X, _Float16* __restrict__ Y) {
  const size_t i = (size_t)blockIdx.x * blockDim.x + threadIdx.x;
  const float4 x = ((const float4*)X)[i];
  v4h y;
  y[0] = (_Float16)x.x; y[1] = (_Float16)x.y;
  y[2] = (_Float16)x.z; y[3] = (_Float16)x.w;
  ((v4h*)Y)[i] = y;
}

// ---------------------------------------------------------------------------
// Pre-pass 2: W[K,N] f32 -> Wt[N,K] f16 (transpose + convert; <=2MB matrices)
// ---------------------------------------------------------------------------
__global__ __launch_bounds__(256) void hm_transpose_cvt(
    const float* __restrict__ W, _Float16* __restrict__ Wt, int K, int N) {
  const size_t idx = (size_t)blockIdx.x * blockDim.x + threadIdx.x;
  const int k = (int)(idx / N);
  const int n = (int)(idx % N);
  Wt[(size_t)n * K + k] = (_Float16)W[idx];
}

// ---------------------------------------------------------------------------
// GEMM: C[M,N] = A[M,K](f16) @ Bt[N,K]^T(f16) + bias; wave tile 32x64,
// 8 wmma / 32-K step.  outF16: write f16 (for Wh/Qh); zeroS: rows with
// row%zeroS==0 forced to zero (t==0 rule of the output projection).
// ---------------------------------------------------------------------------
__global__ __launch_bounds__(128) void hm_gemm(
    const _Float16* __restrict__ A, const _Float16* __restrict__ Bt,
    const float* __restrict__ bias, void* __restrict__ Cout,
    int K, int N, int zeroS, int outF16) {
  const int wave  = threadIdx.x >> 5;
  const int lane  = threadIdx.x & 31;
  const int mbase = blockIdx.x * 32;
  const int nbase = (blockIdx.y * 4 + wave) * 64;
  const int klo   = (lane >> 4) << 3;
  const int r0    = mbase + (lane & 15);
  const int r1    = r0 + 16;

  v8f acc[2][4] = {};
  for (int k0 = 0; k0 < K; k0 += 32) {
    const _Float16* a0p = A + (size_t)r0 * K + (k0 + klo);
    const _Float16* a1p = A + (size_t)r1 * K + (k0 + klo);
    __builtin_prefetch(a0p + 32, 0, 3);           // global_prefetch_b8
    v16h a0 = frag16(a0p);
    v16h a1 = frag16(a1p);
#pragma unroll
    for (int nt = 0; nt < 4; ++nt) {
      const int col = nbase + nt * 16 + (lane & 15);
      v16h b = frag16(Bt + (size_t)col * K + (k0 + klo));
      acc[0][nt] = __builtin_amdgcn_wmma_f32_16x16x32_f16(
          false, a0, false, b, (short)0, acc[0][nt], false, false);
      acc[1][nt] = __builtin_amdgcn_wmma_f32_16x16x32_f16(
          false, a1, false, b, (short)0, acc[1][nt], false, false);
    }
  }
  const int rowoff = (lane >> 4) * 8;             // C/D: lanes 16-31 -> rows+8
#pragma unroll
  for (int mt = 0; mt < 2; ++mt) {
#pragma unroll
    for (int nt = 0; nt < 4; ++nt) {
      const int col = nbase + nt * 16 + (lane & 15);
      const float bv = bias[col];
#pragma unroll
      for (int r = 0; r < 8; ++r) {
        const int orow = mbase + mt * 16 + rowoff + r;
        float v = acc[mt][nt][r] + bv;
        if (zeroS && (orow % zeroS) == 0) v = 0.0f;
        if (outF16) ((_Float16*)Cout)[(size_t)orow * N + col] = (_Float16)v;
        else        ((float*)Cout)[(size_t)orow * N + col] = v;
      }
    }
  }
}

// ---------------------------------------------------------------------------
// Window-64 attention, one wave32 per 16-query tile (tiles never straddle a
// batch since S%16==0).  Phase 1: S = Q*W^T via WMMA over 5 key tiles
// (keys t0-64 .. t0+15; invalid keys masked).  Masked softmax in the C/D
// register layout.  Phase 2: O = P*W in VALU, P broadcast through LDS.
// ---------------------------------------------------------------------------
__global__ __launch_bounds__(256) void hm_attn(
    const _Float16* __restrict__ Wh, const _Float16* __restrict__ Qh,
    _Float16* __restrict__ Rh, int S, int D, float scale) {
  __shared__ float P_lds[8][16][80];
  __shared__ float Linv_lds[8][16];

  const int wv    = threadIdx.x >> 5;
  const int lane  = threadIdx.x & 31;
  const int c     = lane & 15;
  const int half  = lane >> 4;
  const int klo   = half << 3;
  const int tile  = blockIdx.x * 8 + wv;     // 16-row tile id
  const int qrow0 = tile * 16;               // global row base = b*S + t0
  const int t0    = qrow0 % S;

  // ---- Phase 1: scores via WMMA -------------------------------------------
  v8f sacc[5] = {};
  const int aRow = qrow0 + c;                // A-frag query row
  int keyRow[5];                             // B-frag key rows (clamped)
#pragma unroll
  for (int kt = 0; kt < 5; ++kt) {
    int j = t0 - 64 + kt * 16 + c;           // key timestep for this lane
    if (j < 0) j = 0;                        // clamp (masked later)
    keyRow[kt] = (qrow0 - t0) + j;           // b*S + j
  }
  for (int k0 = 0; k0 < D; k0 += 32) {
    v16h a = frag16(Qh + (size_t)aRow * D + (k0 + klo));
#pragma unroll
    for (int kt = 0; kt < 5; ++kt) {
      v16h b = frag16(Wh + (size_t)keyRow[kt] * D + (k0 + klo));
      sacc[kt] = __builtin_amdgcn_wmma_f32_16x16x32_f16(
          false, a, false, b, (short)0, sacc[kt], false, false);
    }
  }

  // ---- masked softmax (row q = t0 + r + 8*half; key j = t0-64+kt*16+c) ----
#pragma unroll
  for (int r = 0; r < 8; ++r) {
    const int t = t0 + r + 8 * half;
    float sv[5];
    float mr = -3.0e30f;
#pragma unroll
    for (int kt = 0; kt < 5; ++kt) {
      const int j = t0 - 64 + kt * 16 + c;
      const bool valid = (j >= 0) && (j >= t - 64) && (j < t);
      const float s = valid ? sacc[kt][r] * scale : -3.0e30f;
      sv[kt] = s;
      mr = fmaxf(mr, s);
    }
#pragma unroll
    for (int off = 1; off < 16; off <<= 1) mr = fmaxf(mr, __shfl_xor(mr, off, 32));
    float lr = 0.0f;
    float pv[5];
#pragma unroll
    for (int kt = 0; kt < 5; ++kt) {
      pv[kt] = (sv[kt] > -1.0e29f) ? __expf(sv[kt] - mr) : 0.0f; // masked -> 0
      lr += pv[kt];
    }
#pragma unroll
    for (int off = 1; off < 16; off <<= 1) lr += __shfl_xor(lr, off, 32);
    const int row = r + 8 * half;
#pragma unroll
    for (int kt = 0; kt < 5; ++kt) P_lds[wv][row][kt * 16 + c] = pv[kt];
    if (c == 0) Linv_lds[wv][row] = (lr > 0.0f) ? (1.0f / lr) : 0.0f;
  }
  // same-wave LDS RAW: hardware DScnt ordering suffices (no cross-wave reads)

  // ---- Phase 2: O = P*W (lane owns a 16-float chunk of D) -----------------
  const int jlo = (t0 >= 64) ? 0 : (64 - t0);  // first in-range key slot
  for (int qc = 0; qc < 16; qc += 8) {
    float acc[8][16] = {};
    for (int j = jlo; j < 80; ++j) {
      const _Float16* wp = Wh + (size_t)(qrow0 - 64 + j) * D + lane * 16;
      v8h w0 = *(const v8h*)wp;
      v8h w1 = *(const v8h*)(wp + 8);
      float wf[16];
#pragma unroll
      for (int i = 0; i < 8; ++i) { wf[i] = (float)w0[i]; wf[8 + i] = (float)w1[i]; }
#pragma unroll
      for (int qq = 0; qq < 8; ++qq) {
        const float p = P_lds[wv][qc + qq][j];
#pragma unroll
        for (int i = 0; i < 16; ++i) acc[qq][i] = fmaf(p, wf[i], acc[qq][i]);
      }
    }
#pragma unroll
    for (int qq = 0; qq < 8; ++qq) {
      const float inv = Linv_lds[wv][qc + qq];
      v8h o0, o1;
#pragma unroll
      for (int i = 0; i < 8; ++i) {
        o0[i] = (_Float16)(acc[qq][i] * inv);
        o1[i] = (_Float16)(acc[qq][8 + i] * inv);
      }
      _Float16* rp = Rh + (size_t)(qrow0 + qc + qq) * D + lane * 16;
      *(v8h*)rp       = o0;
      *(v8h*)(rp + 8) = o1;
    }
  }
}

// ---------------------------------------------------------------------------
extern "C" void kernel_launch(void* const* d_in, const int* in_sizes, int n_in,
                              void* d_out, int out_size, void* d_ws, size_t ws_size,
                              hipStream_t stream) {
  constexpr int B = 8, S = 2048, E = 1024, D = 512;
  const int M = B * S;
  const float* emb = (const float*)d_in[0];
  const float* Ww  = (const float*)d_in[1];
  const float* bw  = (const float*)d_in[2];
  const float* Wq  = (const float*)d_in[3];
  const float* bq  = (const float*)d_in[4];
  const float* Wo  = (const float*)d_in[5];
  const float* bo  = (const float*)d_in[6];
  float* out = (float*)d_out;

  char* ws = (char*)d_ws;
  size_t off = 0;
  auto take = [&](size_t bytes) {
    char* p = ws + off;
    off += (bytes + 255) & ~(size_t)255;
    return p;
  };
  _Float16* embh = (_Float16*)take((size_t)M * E * 2);  // emb f16
  _Float16* Wwt  = (_Float16*)take((size_t)E * D * 2);  // [D,E] f16
  _Float16* Wqt  = (_Float16*)take((size_t)E * D * 2);  // [D,E] f16
  _Float16* Wot  = (_Float16*)take((size_t)D * E * 2);  // [E,D] f16
  _Float16* Wh   = (_Float16*)take((size_t)M * D * 2);  // write vals f16
  _Float16* Qh   = (_Float16*)take((size_t)M * D * 2);  // queries f16
  _Float16* Rh   = (_Float16*)take((size_t)M * D * 2);  // retrieved f16

  const float scale = 0.044194173824159216f;            // 512^-0.5

  hm_cvt_f16<<<(M * (E / 4)) / 256, 256, 0, stream>>>(emb, embh);
  hm_transpose_cvt<<<(E * D) / 256, 256, 0, stream>>>(Ww, Wwt, E, D);
  hm_transpose_cvt<<<(E * D) / 256, 256, 0, stream>>>(Wq, Wqt, E, D);
  hm_transpose_cvt<<<(D * E) / 256, 256, 0, stream>>>(Wo, Wot, D, E);

  dim3 gIn(M / 32, D / 256);
  hm_gemm<<<gIn, 128, 0, stream>>>(embh, Wwt, bw, Wh, E, D, 0, 1);
  hm_gemm<<<gIn, 128, 0, stream>>>(embh, Wqt, bq, Qh, E, D, 0, 1);

  hm_attn<<<M / (16 * 8), 256, 0, stream>>>(Wh, Qh, Rh, S, D, scale);

  dim3 gOut(M / 32, E / 256);
  hm_gemm<<<gOut, 128, 0, stream>>>(Rh, Wot, bo, out, D, E, S, 0);
}
